// GGNN_72215580115133
// MI455X (gfx1250) — compile-verified, matrix operand order checked by reading, not run on hardware
//
#include <hip/hip_runtime.h>
#include <hip/hip_bf16.h>

typedef __attribute__((ext_vector_type(16))) __bf16 v16bf;
typedef __attribute__((ext_vector_type(8)))  __bf16 v8bf;
typedef __attribute__((ext_vector_type(8)))  float  v8f;
typedef __attribute__((ext_vector_type(4)))  int    v4i;

#define AS1 __attribute__((address_space(1)))
#define AS3 __attribute__((address_space(3)))

#define GN 10000
#define GF 80000
#define GE 160000

// ---------------------------------------------------------------------------
// Weight packing: f32 [OUT][IN] row-major -> bf16 WMMA B-fragments.
// Fragment (kt,nt) holds B[k][n] for k in [kt*32,kt*32+32), n in [nt*16,nt*16+16),
// laid out so each lane's 16 bf16 are two contiguous 16B chunks (b128 loads).
// ---------------------------------------------------------------------------
__global__ void pack_w_kernel(const float* __restrict__ w, __bf16* __restrict__ out,
                              int OUT, int IN, int ktiles, int ntiles) {
  int tid = blockIdx.x * blockDim.x + threadIdx.x;
  int total = ktiles * ntiles * 512;
  if (tid >= total) return;
  int j  = tid & 15;
  int l  = (tid >> 4) & 31;
  int fr = tid >> 9;
  int nt = fr % ntiles;
  int kt = fr / ntiles;
  int n  = nt * 16 + (l & 15);
  int kh = l >> 4;
  int k  = kt * 32 + ((j < 8) ? (kh * 8 + j) : (16 + kh * 8 + (j - 8)));
  float v = (k < IN && n < OUT) ? w[(size_t)n * IN + k] : 0.0f;
  out[tid] = (__bf16)v;
}

// ---------------------------------------------------------------------------
// Edge-message MLP: per-block async-DMA weight staging into LDS, then each
// wave grid-strides over 16-edge tiles: gather -> 3x bf16-WMMA layers -> fused
// f32 atomic segment-sum. Wave-private LDS activation slices (no barriers in
// the steady-state loop; LDS ops are in-order within a wave).
// LDS-bound to 1 workgroup/WGP -> give each wave a big VGPR budget (no spills).
// ---------------------------------------------------------------------------
#define WBYTES  90112         // packed weights of one MLP: (40+32+16)*1024 B
#define WCHUNKS 5632          // WBYTES / 16
#define W2OFF   (40 * 512)    // element offsets into the packed-weight block
#define W3OFF   (72 * 512)
#define SLICE   9472          // 5120 (X) + 4096 (H) + 64 (SEG) + pad

__global__ void __launch_bounds__(256, 1)
edge_mlp_kernel(const int* __restrict__ rows, const int* __restrict__ cols,
                const float* __restrict__ sRow, const float* __restrict__ sCol,
                const float* __restrict__ feat,
                const __bf16* __restrict__ pwAll,
                const float* __restrict__ b1, const float* __restrict__ b2,
                const float* __restrict__ b3,
                float* __restrict__ nmOut, int nEdges) {
  extern __shared__ char smem[];
  __bf16* WB    = (__bf16*)smem;                // staged packed weights (90 KB)
  char*   slice = smem + WBYTES + (threadIdx.x >> 5) * SLICE;
  __bf16* X   = (__bf16*)slice;                 // inputs (16x160), later h2 (16x128)
  __bf16* H   = (__bf16*)(slice + 5120);        // h1 (16x128)
  int*    SEG = (int*)(slice + 5120 + 4096);    // destination segment per edge row

  const int wave = threadIdx.x >> 5;
  const int lane = threadIdx.x & 31;
  const int r    = lane & 15;
  const int kh   = lane >> 4;

  // ---- stage packed weights into LDS once per block (async DMA path) ----
  {
    const int t = threadIdx.x;
#if __has_builtin(__builtin_amdgcn_global_load_async_to_lds_b128)
    for (int c = t; c < WCHUNKS; c += 256) {
      __builtin_amdgcn_global_load_async_to_lds_b128(
          (AS1 v4i*)(pwAll + (size_t)c * 8),
          (AS3 v4i*)(WB + (size_t)c * 8), 0, 0);
    }
#if __has_builtin(__builtin_amdgcn_s_wait_asynccnt)
    __builtin_amdgcn_s_wait_asynccnt(0);
#else
    asm volatile("s_wait_asynccnt 0" ::: "memory");
#endif
#else
    for (int c = t; c < WCHUNKS; c += 256)
      ((v8bf*)WB)[c] = ((const v8bf*)pwAll)[c];
#endif
    __syncthreads();
  }

  auto loadA = [&](const __bf16* base, int stride, int kt) -> v16bf {
    v8bf lo = *(const v8bf*)(base + r * stride + kt * 32 + kh * 8);
    v8bf hi = *(const v8bf*)(base + r * stride + kt * 32 + 16 + kh * 8);
    v16bf o;
#pragma unroll
    for (int i = 0; i < 8; ++i) { o[i] = lo[i]; o[i + 8] = hi[i]; }
    return o;
  };
  auto loadB = [&](const __bf16* wb, int frag) -> v16bf {
    v8bf lo = *(const v8bf*)(wb + (size_t)frag * 512 + lane * 16);
    v8bf hi = *(const v8bf*)(wb + (size_t)frag * 512 + lane * 16 + 8);
    v16bf o;
#pragma unroll
    for (int i = 0; i < 8; ++i) { o[i] = lo[i]; o[i + 8] = hi[i]; }
    return o;
  };

  const int nTiles = nEdges >> 4;
  for (int tile = blockIdx.x * 8 + wave; tile < nTiles; tile += (int)gridDim.x * 8) {
    const int e0 = tile * 16;
    const int e  = e0 + r;

    // prefetch next tile's edge indices (global_prefetch_b8)
    const int eNext = e0 + ((int)gridDim.x << 7);
    if (eNext < nEdges) {
      __builtin_prefetch(&rows[eNext], 0, 1);
      __builtin_prefetch(&cols[eNext], 0, 1);
    }

    // ---- gather: concat(sRow[row], sCol[col], feat[row], feat[col]) -> bf16, pad 160
    if (lane < 16) {
      const int ri = rows[e];
      const float* src = sRow + (size_t)ri * 64;
      for (int k = 0; k < 64; ++k) X[r * 160 + k] = (__bf16)src[k];
      for (int j = 0; j < 4; ++j)  X[r * 160 + 128 + j] = (__bf16)feat[(size_t)ri * 4 + j];
    } else {
      const int ci = cols[e];
      const float* src = sCol + (size_t)ci * 64;
      for (int k = 0; k < 64; ++k) X[r * 160 + 64 + k] = (__bf16)src[k];
      for (int j = 0; j < 4; ++j)  X[r * 160 + 132 + j] = (__bf16)feat[(size_t)ci * 4 + j];
      for (int j = 136; j < 160; ++j) X[r * 160 + j] = (__bf16)0.0f;
      SEG[r] = ci;
    }
    // wave-private slice; DS ops within one wave are in order -> no barrier.

    // ---- layer 1: [16x160] @ [160x128], ReLU -> H
    v16bf a1[5];
#pragma unroll
    for (int kt = 0; kt < 5; ++kt) a1[kt] = loadA(X, 160, kt);
#pragma unroll
    for (int nt = 0; nt < 8; ++nt) {
      float bias = b1[nt * 16 + (lane & 15)];
      v8f acc;
#pragma unroll
      for (int i = 0; i < 8; ++i) acc[i] = bias;
#pragma unroll
      for (int kt = 0; kt < 5; ++kt) {
        v16bf b = loadB(WB, kt * 8 + nt);
        acc = __builtin_amdgcn_wmma_f32_16x16x32_bf16(false, a1[kt], false, b,
                                                      (short)0, acc, false, false);
      }
#pragma unroll
      for (int i = 0; i < 8; ++i) {
        float v = acc[i] > 0.0f ? acc[i] : 0.0f;
        H[(i + kh * 8) * 128 + nt * 16 + (lane & 15)] = (__bf16)v;
      }
    }

    // ---- layer 2: [16x128] @ [128x128], ReLU -> X (reuse)
    v16bf a2[4];
#pragma unroll
    for (int kt = 0; kt < 4; ++kt) a2[kt] = loadA(H, 128, kt);
#pragma unroll
    for (int nt = 0; nt < 8; ++nt) {
      float bias = b2[nt * 16 + (lane & 15)];
      v8f acc;
#pragma unroll
      for (int i = 0; i < 8; ++i) acc[i] = bias;
#pragma unroll
      for (int kt = 0; kt < 4; ++kt) {
        v16bf b = loadB(WB + W2OFF, kt * 8 + nt);
        acc = __builtin_amdgcn_wmma_f32_16x16x32_bf16(false, a2[kt], false, b,
                                                      (short)0, acc, false, false);
      }
#pragma unroll
      for (int i = 0; i < 8; ++i) {
        float v = acc[i] > 0.0f ? acc[i] : 0.0f;
        X[(i + kh * 8) * 128 + nt * 16 + (lane & 15)] = (__bf16)v;
      }
    }

    // ---- layer 3: [16x128] @ [128x64] -> fused f32 atomic segment-sum
    v16bf a3[4];
#pragma unroll
    for (int kt = 0; kt < 4; ++kt) a3[kt] = loadA(X, 128, kt);
    int segs[8];
#pragma unroll
    for (int i = 0; i < 8; ++i) segs[i] = SEG[i + kh * 8];
#pragma unroll
    for (int nt = 0; nt < 4; ++nt) {
      float bias = b3[nt * 16 + (lane & 15)];
      v8f acc;
#pragma unroll
      for (int i = 0; i < 8; ++i) acc[i] = bias;
#pragma unroll
      for (int kt = 0; kt < 4; ++kt) {
        v16bf b = loadB(WB + W3OFF, kt * 4 + nt);
        acc = __builtin_amdgcn_wmma_f32_16x16x32_bf16(false, a3[kt], false, b,
                                                      (short)0, acc, false, false);
      }
#pragma unroll
      for (int i = 0; i < 8; ++i) {
        float* dst = &nmOut[(size_t)segs[i] * 64 + nt * 16 + (lane & 15)];
        __hip_atomic_fetch_add(dst, acc[i], __ATOMIC_RELAXED, __HIP_MEMORY_SCOPE_AGENT);
      }
    }
  }
}

// ---------------------------------------------------------------------------
// GRU cell: weights staged in LDS once per block, 64 rows per block.
// blockDim = 192 (one thread per gate output j in [0,192)).
// ---------------------------------------------------------------------------
__global__ void __launch_bounds__(192, 1)
gru_kernel(const float* __restrict__ x, float* __restrict__ h,
           const float* __restrict__ wih, const float* __restrict__ whh,
           const float* __restrict__ bih, const float* __restrict__ bhh,
           int nRows) {
  extern __shared__ char smem[];
  float* Wih = (float*)smem;        // 192*64
  float* Whh = Wih + 192 * 64;      // 192*64
  float* Bih = Whh + 192 * 64;      // 192
  float* Bhh = Bih + 192;           // 192
  float* Xs  = Bhh + 192;           // 64
  float* Hs  = Xs + 64;             // 64
  float* Gi  = Hs + 64;             // 192
  float* Gh  = Gi + 192;            // 192
  const int t = threadIdx.x;
  for (int i = t; i < 192 * 64; i += 192) { Wih[i] = wih[i]; Whh[i] = whh[i]; }
  if (t < 192) { Bih[t] = bih[t]; Bhh[t] = bhh[t]; }
  __syncthreads();

  const int row0 = blockIdx.x * 64;
  for (int rr = 0; rr < 64; ++rr) {
    const int row = row0 + rr;
    if (row >= nRows) break;
    if (t < 64)        Xs[t]      = x[(size_t)row * 64 + t];
    else if (t < 128)  Hs[t - 64] = h[(size_t)row * 64 + (t - 64)];
    __syncthreads();
    float gi = Bih[t], gh = Bhh[t];
    const float* wi = &Wih[t * 64];
    const float* wh = &Whh[t * 64];
    for (int m = 0; m < 64; ++m) { gi += Xs[m] * wi[m]; gh += Hs[m] * wh[m]; }
    Gi[t] = gi; Gh[t] = gh;
    __syncthreads();
    if (t < 64) {
      float rg = 1.0f / (1.0f + __expf(-(Gi[t] + Gh[t])));
      float zg = 1.0f / (1.0f + __expf(-(Gi[64 + t] + Gh[64 + t])));
      float nn = tanhf(Gi[128 + t] + rg * Gh[128 + t]);
      h[(size_t)row * 64 + t] = (1.0f - zg) * nn + zg * Hs[t];
    }
    __syncthreads();
  }
}

// ---------------------------------------------------------------------------
// Readout MLP + softmax: weights in LDS, 64 rows per block, blockDim = 128.
// ---------------------------------------------------------------------------
__global__ void __launch_bounds__(128, 1)
readout_kernel(const float* __restrict__ h,
               const float* __restrict__ w1, const float* __restrict__ b1,
               const float* __restrict__ w2, const float* __restrict__ b2,
               const float* __restrict__ w3, const float* __restrict__ b3,
               float* __restrict__ out, int nRows) {
  extern __shared__ char smem[];
  float* W1 = (float*)smem;         // 128*64
  float* W2 = W1 + 128 * 64;        // 128*128
  float* W3 = W2 + 128 * 128;       // 2*128 (pad 256)
  float* B1 = W3 + 256;             // 128
  float* B2 = B1 + 128;             // 128
  float* B3 = B2 + 128;             // 4
  float* Hs = B3 + 4;               // 64
  float* H1 = Hs + 64;              // 128
  float* H2 = H1 + 128;             // 128
  const int t = threadIdx.x;
  for (int i = t; i < 128 * 64;  i += 128) W1[i] = w1[i];
  for (int i = t; i < 128 * 128; i += 128) W2[i] = w2[i];
  for (int i = t; i < 256;       i += 128) W3[i] = w3[i];
  B1[t] = b1[t];
  B2[t] = b2[t];
  if (t < 2) B3[t] = b3[t];
  __syncthreads();

  const int row0 = blockIdx.x * 64;
  for (int rr = 0; rr < 64; ++rr) {
    const int row = row0 + rr;
    if (row >= nRows) break;
    if (t < 64) Hs[t] = h[(size_t)row * 64 + t];
    __syncthreads();
    float a = B1[t];
    for (int k = 0; k < 64; ++k) a += Hs[k] * W1[t * 64 + k];
    H1[t] = a > 0.0f ? a : 0.0f;
    __syncthreads();
    a = B2[t];
    for (int k = 0; k < 128; ++k) a += H1[k] * W2[t * 128 + k];
    H2[t] = a > 0.0f ? a : 0.0f;
    __syncthreads();
    if (t == 0) {
      float l0 = B3[0], l1 = B3[1];
      for (int k = 0; k < 128; ++k) { l0 += H2[k] * W3[k]; l1 += H2[k] * W3[128 + k]; }
      float mx = l0 > l1 ? l0 : l1;
      float e0 = __expf(l0 - mx), e1 = __expf(l1 - mx);
      float inv = 1.0f / (e0 + e1);
      out[(size_t)row * 2 + 0] = e0 * inv;
      out[(size_t)row * 2 + 1] = e1 * inv;
    }
    __syncthreads();
  }
}

// ---------------------------------------------------------------------------
extern "C" void kernel_launch(void* const* d_in, const int* in_sizes, int n_in,
                              void* d_out, int out_size, void* d_ws, size_t ws_size,
                              hipStream_t stream) {
  (void)in_sizes; (void)n_in; (void)out_size; (void)ws_size;
  const int*   f2v_row  = (const int*)d_in[0];
  const int*   f2v_col  = (const int*)d_in[1];
  const int*   v2f_row  = (const int*)d_in[2];
  const int*   v2f_col  = (const int*)d_in[3];
  const float* f2v_feat = (const float*)d_in[4];
  const float* v2f_feat = (const float*)d_in[5];
  const float* f2v_w1 = (const float*)d_in[6];  const float* f2v_b1 = (const float*)d_in[7];
  const float* f2v_w2 = (const float*)d_in[8];  const float* f2v_b2 = (const float*)d_in[9];
  const float* f2v_w3 = (const float*)d_in[10]; const float* f2v_b3 = (const float*)d_in[11];
  const float* v2f_w1 = (const float*)d_in[12]; const float* v2f_b1 = (const float*)d_in[13];
  const float* v2f_w2 = (const float*)d_in[14]; const float* v2f_b2 = (const float*)d_in[15];
  const float* v2f_w3 = (const float*)d_in[16]; const float* v2f_b3 = (const float*)d_in[17];
  const float* gf_wih = (const float*)d_in[18]; const float* gf_whh = (const float*)d_in[19];
  const float* gf_bih = (const float*)d_in[20]; const float* gf_bhh = (const float*)d_in[21];
  const float* gv_wih = (const float*)d_in[22]; const float* gv_whh = (const float*)d_in[23];
  const float* gv_bih = (const float*)d_in[24]; const float* gv_bhh = (const float*)d_in[25];
  const float* ro_w1 = (const float*)d_in[26]; const float* ro_b1 = (const float*)d_in[27];
  const float* ro_w2 = (const float*)d_in[28]; const float* ro_b2 = (const float*)d_in[29];
  const float* ro_w3 = (const float*)d_in[30]; const float* ro_b3 = (const float*)d_in[31];

  // ---- workspace carve-out (~44 MiB) ----
  char* ws = (char*)d_ws;
  float* var_h  = (float*)ws; ws += (size_t)GN * 64 * 4;
  float* fac_h  = (float*)ws; ws += (size_t)GF * 64 * 4;
  float* nm_var = (float*)ws; ws += (size_t)GN * 64 * 4;
  float* nm_fac = (float*)ws; ws += (size_t)GF * 64 * 4;
  __bf16* pf_w1 = (__bf16*)ws; ws += 40 * 512 * 2;   // 5 kt x 8 nt (contiguous w1,w2,w3)
  __bf16* pf_w2 = (__bf16*)ws; ws += 32 * 512 * 2;   // 4 x 8
  __bf16* pf_w3 = (__bf16*)ws; ws += 16 * 512 * 2;   // 4 x 4
  __bf16* pv_w1 = (__bf16*)ws; ws += 40 * 512 * 2;
  __bf16* pv_w2 = (__bf16*)ws; ws += 32 * 512 * 2;
  __bf16* pv_w3 = (__bf16*)ws; ws += 16 * 512 * 2;

  (void)hipMemsetAsync(var_h, 0, (size_t)GN * 64 * 4, stream);
  (void)hipMemsetAsync(fac_h, 0, (size_t)GF * 64 * 4, stream);

  // ---- pack weights into WMMA B-fragment layout (bf16) ----
  auto pack = [&](const float* w, __bf16* o, int OUT, int IN, int kt, int nt) {
    int total = kt * nt * 512;
    pack_w_kernel<<<(total + 255) / 256, 256, 0, stream>>>(w, o, OUT, IN, kt, nt);
  };
  pack(f2v_w1, pf_w1, 128, 136, 5, 8);
  pack(f2v_w2, pf_w2, 128, 128, 4, 8);
  pack(f2v_w3, pf_w3,  64, 128, 4, 4);
  pack(v2f_w1, pv_w1, 128, 136, 5, 8);
  pack(v2f_w2, pv_w2, 128, 128, 4, 8);
  pack(v2f_w3, pv_w3,  64, 128, 4, 4);

  const int    EDGE_BLOCKS = 256;                 // grid-stride over 10000 tiles
  const size_t EDGE_SHM = WBYTES + 8 * SLICE;     // 165888 B per workgroup
  const size_t GRU_SHM  = (size_t)(192 * 64 * 2 + 192 * 2 + 64 * 2 + 192 * 2) * 4;
  const size_t RO_SHM   = (size_t)(128 * 64 + 128 * 128 + 256 + 128 + 128 + 4 + 64 + 128 + 128) * 4;

  for (int step = 0; step < 5; ++step) {
    // fac -> var
    (void)hipMemsetAsync(nm_var, 0, (size_t)GN * 64 * 4, stream);
    edge_mlp_kernel<<<EDGE_BLOCKS, 256, EDGE_SHM, stream>>>(
        f2v_row, f2v_col, fac_h, var_h, f2v_feat,
        pf_w1, f2v_b1, f2v_b2, f2v_b3, nm_var, GE);
    gru_kernel<<<(GN + 63) / 64, 192, GRU_SHM, stream>>>(
        nm_var, var_h, gf_wih, gf_whh, gf_bih, gf_bhh, GN);
    // var -> fac
    (void)hipMemsetAsync(nm_fac, 0, (size_t)GF * 64 * 4, stream);
    edge_mlp_kernel<<<EDGE_BLOCKS, 256, EDGE_SHM, stream>>>(
        v2f_row, v2f_col, var_h, fac_h, v2f_feat,
        pv_w1, v2f_b1, v2f_b2, v2f_b3, nm_fac, GE);
    gru_kernel<<<(GF + 63) / 64, 192, GRU_SHM, stream>>>(
        nm_fac, fac_h, gv_wih, gv_whh, gv_bih, gv_bhh, GF);
  }

  readout_kernel<<<(GN + 63) / 64, 128, RO_SHM, stream>>>(
      var_h, ro_w1, ro_b1, ro_w2, ro_b2, ro_w3, ro_b3, (float*)d_out, GN);
}
